// NaiveAttentionBasedEncoder_41506563949126
// MI455X (gfx1250) — matrix operand order repaired and checked by reading, not compiled, not run on hardware
//
#include <hip/hip_runtime.h>
#include <hip/hip_bf16.h>
#include <math.h>

// Graph-attention encoder for MI455X (gfx1250, wave32).
// GEMMs: v_wmma_f32_16x16x32_bf16; each wave computes a 16x128 output strip
// (8 accumulator tiles, A fragment loaded once per K-step). The 128x128 bf16
// weight matrix is staged into LDS (32 KB) per block with
// global_load_async_to_lds_b128 (ASYNCcnt) + s_wait_asynccnt + barrier.
// Edge-sized streams are bf16 (memory-bound at 23.3 TB/s, halves traffic);
// node tables h/q (~51 MB) stay L2-resident (192 MB) for per-edge gathers.
// Requires ws_size >= ~576 MB.

#define NN 50000
#define NE 600000
#define DD 128
#define NH 4

typedef __bf16 bf16_t;
typedef __attribute__((ext_vector_type(16))) bf16_t        bf16x16;
typedef __attribute__((ext_vector_type(8)))  float         f32x8;
typedef __attribute__((ext_vector_type(8)))  unsigned int  u32x8;

static __device__ __forceinline__ unsigned short f32_bf16(float f) {
    unsigned int u = __float_as_uint(f);
    unsigned int lsb = (u >> 16) & 1u;          // round-to-nearest-even
    return (unsigned short)((u + 0x7FFFu + lsb) >> 16);
}
static __device__ __forceinline__ float bf16_f32(unsigned short h) {
    return __uint_as_float(((unsigned int)h) << 16);
}

// ---------------------------------------------------------------------------
// f32 -> bf16 stream convert
// ---------------------------------------------------------------------------
__global__ void cvt_f32_to_bf16(const float* __restrict__ in,
                                unsigned short* __restrict__ out, int n) {
    int i = blockIdx.x * blockDim.x + threadIdx.x;
    if (i < n) out[i] = f32_bf16(in[i]);
}

// ---------------------------------------------------------------------------
// Strip WMMA GEMM: C[Mx128] = A[Mx128] * B[128x128]  (row-major, bf16 in,
// f32 accum). One 16-row x 128-col strip per wave (8 WMMA tiles), 8 waves
// per block. B staged once per block into LDS via async load (ASYNCcnt).
// Optional fused epilogue: C += addTab[addIdx[row]]  (xj = h[src] + e).
// ---------------------------------------------------------------------------
template <bool HAS_ADD>
__global__ void __launch_bounds__(256)
wmma_gemm_strip(const unsigned short* __restrict__ A,
                const unsigned short* __restrict__ B,
                float* __restrict__ Cf,
                unsigned short* __restrict__ Cb,
                const float* __restrict__ addTab,
                const int* __restrict__ addIdx,
                int M) {
    constexpr int K = DD, N = DD;
    __shared__ __attribute__((aligned(32))) unsigned short Bsh[DD * DD]; // 32 KB

    // ---- async stage B (128x128 bf16) into LDS: 2048 x 16B, 8 per thread ----
    {
        unsigned ldsbase = (unsigned)(size_t)(void*)&Bsh[0];   // low 32b = LDS offset
        unsigned long long gbase = (unsigned long long)(size_t)B;
        #pragma unroll
        for (int i = 0; i < 8; ++i) {
            unsigned byteoff = (threadIdx.x + i * 256u) * 16u;
            unsigned lds = ldsbase + byteoff;
            asm volatile("global_load_async_to_lds_b128 %0, %1, %2"
                         :: "v"(lds), "v"(byteoff), "s"(gbase) : "memory");
        }
        asm volatile("s_wait_asynccnt 0" ::: "memory");
        __syncthreads();   // all waves staged before any divergent exit
    }

    const int lane = threadIdx.x & 31;
    const int wid  = blockIdx.x * 8 + (threadIdx.x >> 5);
    const int m0   = wid << 4;
    if (m0 >= M) return;                       // wave-uniform (EXEC all-1 for WMMA)

    int rowA = m0 + (lane & 15);
    if (rowA > M - 1) rowA = M - 1;            // clamp tail reads; stores guarded
    const unsigned short* arow = A + (size_t)rowA * K;
    const int ksel = (lane < 16) ? 0 : 8;      // A-frag K half per lane group
    __builtin_prefetch(arow, 0, 1);            // global_prefetch_b8

    f32x8 acc[8];
    #pragma unroll
    for (int t = 0; t < 8; ++t) acc[t] = (f32x8){};

    for (int k0 = 0; k0 < K; k0 += 32) {
        // A 16x32 bf16 fragment: VGPR j<4 -> K = ksel+2j, j>=4 -> 16+ksel+2(j-4)
        u32x8 au;
        #pragma unroll
        for (int j = 0; j < 8; ++j) {
            int kk = k0 + ((j >= 4) ? 16 : 0) + ksel + ((j & 3) << 1);
            au[j] = *(const unsigned int*)(arow + kk);
        }
        bf16x16 afrag = __builtin_bit_cast(bf16x16, au);

        // B fragments from LDS: lane = col (lane&15) of tile t; elements over K in A's order
        const int colB = lane & 15;
        #pragma unroll
        for (int t = 0; t < 8; ++t) {
            bf16x16 bfrag;
            #pragma unroll
            for (int s = 0; s < 16; ++s) {
                int kk = k0 + ((s >= 8) ? 16 : 0) + ksel + (s & 7);
                bfrag[s] = __builtin_bit_cast(bf16_t, Bsh[kk * N + t * 16 + colB]);
            }
            acc[t] = __builtin_amdgcn_wmma_f32_16x16x32_bf16(
                false, afrag, false, bfrag,
                (short)0, acc[t], false, false);
        }
    }

    // C/D layout: lane<16 -> rows m0+0..7, lane>=16 -> rows m0+8..15
    const int colL  = lane & 15;
    const int rbase = m0 + ((lane < 16) ? 0 : 8);
    int gidx[8];
    if (HAS_ADD) {
        #pragma unroll
        for (int r = 0; r < 8; ++r) {
            int row = rbase + r;
            gidx[r] = addIdx[row < M ? row : 0];
        }
    }
    #pragma unroll
    for (int t = 0; t < 8; ++t) {
        int col = t * 16 + colL;
        #pragma unroll
        for (int r = 0; r < 8; ++r) {
            int row = rbase + r;
            if (row < M) {
                float v = acc[t][r];
                if (HAS_ADD) v += addTab[(size_t)gidx[r] * N + col]; // h[src] (L2)
                if (Cf) Cf[(size_t)row * N + col] = v;
                if (Cb) Cb[(size_t)row * N + col] = f32_bf16(v);
            }
        }
    }
}

// ---------------------------------------------------------------------------
// Per-(edge,head) score = scale * dot(q[dst[e]][h], k[e][h]) + segment max
// ---------------------------------------------------------------------------
__global__ void scores_kernel(const unsigned short* __restrict__ Kb,
                              const float* __restrict__ Qf,
                              const int* __restrict__ dst,
                              float* __restrict__ sc,
                              float* __restrict__ mbuf) {
    int gid = blockIdx.x * blockDim.x + threadIdx.x;
    if (gid >= NE * NH) return;
    int e = gid >> 2, h = gid & 3;
    int node = dst[e];
    const unsigned short* kp = Kb + (size_t)e * DD + h * 32;
    const float* qp = Qf + (size_t)node * DD + h * 32;
    float s = 0.f;
    #pragma unroll
    for (int i = 0; i < 32; i += 2) {
        unsigned int pk = *(const unsigned int*)(kp + i);
        s += bf16_f32((unsigned short)(pk & 0xffffu)) * qp[i];
        s += bf16_f32((unsigned short)(pk >> 16))     * qp[i + 1];
    }
    s *= 0.17677669529663687f;                 // 1/sqrt(32)
    sc[gid] = s;
    // monotone float atomic-max: int-max for >=0, uint-min for <0 (init = -inf)
    float* maddr = mbuf + (size_t)node * NH + h;
    if (!(s < 0.f)) atomicMax((int*)maddr, __float_as_int(s));
    else            atomicMin((unsigned int*)maddr, __float_as_uint(s));
}

__global__ void softmax_p_kernel(const int* __restrict__ dst,
                                 const float* __restrict__ mbuf,
                                 float* __restrict__ sc,
                                 float* __restrict__ dbuf) {
    int gid = blockIdx.x * blockDim.x + threadIdx.x;
    if (gid >= NE * NH) return;
    int e = gid >> 2, h = gid & 3;
    int node = dst[e];
    float p = __expf(sc[gid] - mbuf[(size_t)node * NH + h]);
    sc[gid] = p;                               // overwrite scores with p
    atomicAdd(dbuf + (size_t)node * NH + h, p);
}

// one block per edge, 128 threads = one feature each
__global__ void agg_kernel(const int* __restrict__ dst,
                           const float* __restrict__ sc,
                           const float* __restrict__ dbuf,
                           const unsigned short* __restrict__ Vb,
                           float* __restrict__ agg) {
    int e = blockIdx.x;
    int d = threadIdx.x;
    int h = d >> 5;
    int node = dst[e];
    float alpha = sc[e * NH + h] / (dbuf[(size_t)node * NH + h] + 1e-16f);
    float vv = bf16_f32(Vb[(size_t)e * DD + d]);
    atomicAdd(agg + (size_t)node * DD + d, alpha * vv);
}

__global__ void init_md_kernel(float* __restrict__ m, float* __restrict__ d, int n) {
    int i = blockIdx.x * blockDim.x + threadIdx.x;
    if (i < n) { m[i] = __int_as_float(0xFF800000); d[i] = 0.f; }
}
__global__ void zero_f32_kernel(float* __restrict__ p, int n) {
    int i = blockIdx.x * blockDim.x + threadIdx.x;
    if (i < n) p[i] = 0.f;
}
__global__ void final_kernel(const float* __restrict__ hf,
                             const float* __restrict__ agg,
                             float* __restrict__ out, int n) {
    int i = blockIdx.x * blockDim.x + threadIdx.x;
    if (i < n) out[i] = fmaxf(hf[i] + agg[i], 0.f);
}

// ---------------------------------------------------------------------------
extern "C" void kernel_launch(void* const* d_in, const int* in_sizes, int n_in,
                              void* d_out, int out_size, void* d_ws, size_t ws_size,
                              hipStream_t stream) {
    const float* x  = (const float*)d_in[0];
    const float* ea = (const float*)d_in[1];
    const int*   ei = (const int*)d_in[2];
    const float* Wn = (const float*)d_in[3];
    const float* We = (const float*)d_in[4];
    const float* Wq = (const float*)d_in[5];
    const float* Wk = (const float*)d_in[6];
    const float* Wv = (const float*)d_in[7];
    const int* src = ei;
    const int* dst = ei + NE;
    float* out = (float*)d_out;

    char* ws = (char*)d_ws;
    size_t off = 0;
    auto carve = [&](size_t bytes) -> char* {
        char* p = ws + off;
        off += (bytes + 255) & ~(size_t)255;
        return p;
    };
    unsigned short* Wn_b = (unsigned short*)carve((size_t)DD * DD * 2);
    unsigned short* We_b = (unsigned short*)carve((size_t)DD * DD * 2);
    unsigned short* Wq_b = (unsigned short*)carve((size_t)DD * DD * 2);
    unsigned short* Wk_b = (unsigned short*)carve((size_t)DD * DD * 2);
    unsigned short* Wv_b = (unsigned short*)carve((size_t)DD * DD * 2);
    unsigned short* x_b  = (unsigned short*)carve((size_t)NN * DD * 2);
    unsigned short* ea_b = (unsigned short*)carve((size_t)NE * DD * 2);
    float*          h_f  = (float*)carve((size_t)NN * DD * 4);
    unsigned short* h_b  = (unsigned short*)carve((size_t)NN * DD * 2);
    float*          q_f  = (float*)carve((size_t)NN * DD * 4);
    unsigned short* xj_b = (unsigned short*)carve((size_t)NE * DD * 2);
    unsigned short* v_b  = (unsigned short*)carve((size_t)NE * DD * 2);
    float*          sc   = (float*)carve((size_t)NE * NH * 4);
    float*          mbuf = (float*)carve((size_t)NN * NH * 4);
    float*          dbuf = (float*)carve((size_t)NN * NH * 4);
    float*          aggb = (float*)carve((size_t)NN * DD * 4);
    unsigned short* k_b  = ea_b;   // reuse: edge_attr bf16 dead after xj GEMM

    const int stripsN = (NN + 15) / 16;    // 3125 (exact)
    const int stripsE = (NE + 15) / 16;    // 37500 (exact)
    const int blocksN = (stripsN + 7) / 8; // 8 waves (strips) per block
    const int blocksE = (stripsE + 7) / 8;

    // f32 -> bf16 conversions
    cvt_f32_to_bf16<<<(DD * DD + 255) / 256, 256, 0, stream>>>(Wn, Wn_b, DD * DD);
    cvt_f32_to_bf16<<<(DD * DD + 255) / 256, 256, 0, stream>>>(We, We_b, DD * DD);
    cvt_f32_to_bf16<<<(DD * DD + 255) / 256, 256, 0, stream>>>(Wq, Wq_b, DD * DD);
    cvt_f32_to_bf16<<<(DD * DD + 255) / 256, 256, 0, stream>>>(Wk, Wk_b, DD * DD);
    cvt_f32_to_bf16<<<(DD * DD + 255) / 256, 256, 0, stream>>>(Wv, Wv_b, DD * DD);
    cvt_f32_to_bf16<<<(NN * DD + 255) / 256, 256, 0, stream>>>(x, x_b, NN * DD);
    cvt_f32_to_bf16<<<(NE * DD + 255) / 256, 256, 0, stream>>>(ea, ea_b, NE * DD);

    // h = x @ Wn  (keep f32 for residual/gather + bf16 for next GEMM)
    wmma_gemm_strip<false><<<blocksN, 256, 0, stream>>>(
        x_b, Wn_b, h_f, h_b, nullptr, nullptr, NN);
    // q = h @ Wq
    wmma_gemm_strip<false><<<blocksN, 256, 0, stream>>>(
        h_b, Wq_b, q_f, nullptr, nullptr, nullptr, NN);
    // xj = h[src] + edge_attr @ We   (fused add-gather epilogue)
    wmma_gemm_strip<true><<<blocksE, 256, 0, stream>>>(
        ea_b, We_b, nullptr, xj_b, h_f, src, NE);
    // k = xj @ Wk   (writes into reused ea_b region)
    wmma_gemm_strip<false><<<blocksE, 256, 0, stream>>>(
        xj_b, Wk_b, nullptr, k_b, nullptr, nullptr, NE);
    // v = xj @ Wv
    wmma_gemm_strip<false><<<blocksE, 256, 0, stream>>>(
        xj_b, Wv_b, nullptr, v_b, nullptr, nullptr, NE);

    // segment softmax + aggregation
    init_md_kernel<<<(NN * NH + 255) / 256, 256, 0, stream>>>(mbuf, dbuf, NN * NH);
    zero_f32_kernel<<<(NN * DD + 255) / 256, 256, 0, stream>>>(aggb, NN * DD);
    scores_kernel<<<(NE * NH + 255) / 256, 256, 0, stream>>>(k_b, q_f, dst, sc, mbuf);
    softmax_p_kernel<<<(NE * NH + 255) / 256, 256, 0, stream>>>(dst, mbuf, sc, dbuf);
    agg_kernel<<<NE, DD, 0, stream>>>(dst, sc, dbuf, v_b, aggb);
    final_kernel<<<(NN * DD + 255) / 256, 256, 0, stream>>>(h_f, aggb, out, NN * DD);
}
